// GroupedQueryAttention_23356032156264
// MI455X (gfx1250) — compile-verified, matrix operand order checked by reading, not run on hardware
//
#include <hip/hip_runtime.h>

// ---------------------------------------------------------------------------
// Types for CDNA5 WMMA (wave32)
// ---------------------------------------------------------------------------
typedef __attribute__((ext_vector_type(16))) __bf16 v16bf;
typedef __attribute__((ext_vector_type(8)))  __bf16 v8bf;
typedef __attribute__((ext_vector_type(4)))  __bf16 v4bf;
typedef __attribute__((ext_vector_type(8)))  float  v8f;

#define Bn 2
#define Sn 2048
#define Dn 2048
#define Hn 16
#define Gn 4
#define HDn 128

__device__ __forceinline__ v16bf cat16(v8bf lo, v8bf hi) {
  return __builtin_shufflevector(lo, hi, 0,1,2,3,4,5,6,7,8,9,10,11,12,13,14,15);
}

__device__ __forceinline__ v8bf cvt8(const float4& a, const float4& b) {
  v8bf r;
  r[0] = (__bf16)a.x; r[1] = (__bf16)a.y; r[2] = (__bf16)a.z; r[3] = (__bf16)a.w;
  r[4] = (__bf16)b.x; r[5] = (__bf16)b.y; r[6] = (__bf16)b.z; r[7] = (__bf16)b.w;
  return r;
}

// gfx1250 async global->LDS copy (ASYNCcnt path). vdst = per-lane LDS byte
// offset (low 32 bits of the generic shared pointer == LDS_ADDR per ISA 10.2).
__device__ __forceinline__ void async_copy_b128(const __bf16* gptr, const __bf16* lptr) {
  uint32_t lds_off = (uint32_t)(size_t)lptr;
  asm volatile("global_load_async_to_lds_b128 %0, %1, off"
               :: "v"(lds_off), "v"(gptr) : "memory");
}

// ---------------------------------------------------------------------------
// Precision prepasses: convert activations f32->bf16 (flat) and weights
// f32 [K,N] -> bf16 [N,K] (transposed so GEMM B-fragments are contiguous-K).
// ---------------------------------------------------------------------------
__global__ __launch_bounds__(256) void cvt_bf16_kernel(
    const float* __restrict__ src, __bf16* __restrict__ dst)
{
  const size_t i = ((size_t)blockIdx.x * 256 + threadIdx.x) * 8;
  float4 a = *(const float4*)(src + i);
  float4 b = *(const float4*)(src + i + 4);
  *(v8bf*)(dst + i) = cvt8(a, b);
}

__global__ __launch_bounds__(256) void wtrans_bf16_kernel(
    const float* __restrict__ W, __bf16* __restrict__ Wt, int N, int K)
{
  const size_t c = (size_t)blockIdx.x * 256 + threadIdx.x;  // float4 chunk
  const int k = (int)(c / (size_t)(N / 4));
  const int n = (int)(c % (size_t)(N / 4)) * 4;
  float4 f = *(const float4*)(W + (size_t)k * N + n);
  Wt[(size_t)(n + 0) * K + k] = (__bf16)f.x;
  Wt[(size_t)(n + 1) * K + k] = (__bf16)f.y;
  Wt[(size_t)(n + 2) * K + k] = (__bf16)f.z;
  Wt[(size_t)(n + 3) * K + k] = (__bf16)f.w;
}

// ---------------------------------------------------------------------------
// Pure-bf16 GEMM: C[M,N](f32) = A[M,K](bf16) * Bt[N,K](bf16)^T + bias[N]
// block = 256 threads (8 waves), block tile 64x128, wave tile 32x32, K-step 32.
// Tiles staged by async DMA into double-buffered LDS; WMMA overlaps next fill.
// ---------------------------------------------------------------------------
__global__ __launch_bounds__(256) void gemm_bf16_kernel(
    const __bf16* __restrict__ A, const __bf16* __restrict__ Bt,
    const float* __restrict__ bias, float* __restrict__ C,
    int M, int N, int K)
{
  __shared__ __align__(16) __bf16 lA[2][64 * 32];    // raw [m][k]
  __shared__ __align__(16) __bf16 lB[2][128 * 32];   // raw [n][k]

  const int tid  = threadIdx.x;
  const int lane = tid & 31;
  const int w    = tid >> 5;   // 0..7
  const int wr   = w >> 2;     // 0..1 (M dir)
  const int wc   = w & 3;      // 0..3 (N dir)
  const int halo = lane >> 4;  // 0/1
  const int lm   = lane & 15;
  const int m0   = blockIdx.y * 64;
  const int n0   = blockIdx.x * 128;

  // staging coordinates (8 bf16 = one b128 per async op per lane)
  const int ar = tid >> 2;            // 0..63
  const int ac = (tid & 3) * 8;       // 0,8,16,24

  auto issue = [&](int buf, int k0) {
    async_copy_b128(A + (size_t)(m0 + ar) * K + k0 + ac, &lA[buf][ar * 32 + ac]);
    #pragma unroll
    for (int i = 0; i < 2; ++i) {
      const int c  = tid + i * 256;       // 0..511
      const int nn = c >> 2;              // 0..127
      const int cc = (c & 3) * 8;
      async_copy_b128(Bt + (size_t)(n0 + nn) * K + k0 + cc, &lB[buf][nn * 32 + cc]);
    }
  };

  v8f acc[2][2];
  #pragma unroll
  for (int i = 0; i < 2; ++i)
    #pragma unroll
    for (int j = 0; j < 2; ++j) acc[i][j] = (v8f){};

  const int KT = K / 32;
  issue(0, 0);

  for (int kt = 0; kt < KT; ++kt) {
    const int buf = kt & 1;
    if (kt + 1 < KT) {
      issue(buf ^ 1, (kt + 1) * 32);
      // 3 async instrs just issued for next tile; drain the older batch only.
      asm volatile("s_wait_asynccnt 0x3" ::: "memory");
    } else {
      asm volatile("s_wait_asynccnt 0x0" ::: "memory");
    }
    __syncthreads();  // all waves' DMA for buf complete -> tile visible

    v16bf af[2], bfm[2];
    #pragma unroll
    for (int mt = 0; mt < 2; ++mt) {
      const int row = wr * 32 + mt * 16 + lm;
      v8bf lo = *(const v8bf*)&lA[buf][row * 32 + halo * 8];
      v8bf hi = *(const v8bf*)&lA[buf][row * 32 + halo * 8 + 16];
      af[mt] = cat16(lo, hi);
    }
    #pragma unroll
    for (int nt = 0; nt < 2; ++nt) {
      const int col = wc * 32 + nt * 16 + lm;
      v8bf lo = *(const v8bf*)&lB[buf][col * 32 + halo * 8];
      v8bf hi = *(const v8bf*)&lB[buf][col * 32 + halo * 8 + 16];
      bfm[nt] = cat16(lo, hi);
    }
    #pragma unroll
    for (int mt = 0; mt < 2; ++mt)
      #pragma unroll
      for (int nt = 0; nt < 2; ++nt)
        acc[mt][nt] = __builtin_amdgcn_wmma_f32_16x16x32_bf16(
            false, af[mt], false, bfm[nt], (short)0, acc[mt][nt], false, false);

    __syncthreads();  // protect buf^1 from refill while others still read it
  }

  #pragma unroll
  for (int mt = 0; mt < 2; ++mt)
    #pragma unroll
    for (int nt = 0; nt < 2; ++nt)
      #pragma unroll
      for (int r = 0; r < 8; ++r) {
        const int row = m0 + wr * 32 + mt * 16 + r + halo * 8;
        const int col = n0 + wc * 32 + nt * 16 + lm;
        C[(size_t)row * N + col] = acc[mt][nt][r] + bias[col];
      }
}

// ---------------------------------------------------------------------------
// prep: per (b,s,head) row of 128: optional RMS-norm (q_scale), optional
// interleaved RoPE, write bf16. TRANS=1 writes [B,NH,HD,S] (for V).
// ---------------------------------------------------------------------------
template <bool RMS, bool ROPE, bool TRANS>
__global__ __launch_bounds__(256) void prep_kernel(
    const float* __restrict__ src, __bf16* __restrict__ dst,
    const float* __restrict__ scale, int NH)
{
  const int wid  = blockIdx.x * 8 + (threadIdx.x >> 5);  // ((b*S+s)*NH + h)
  const int lane = threadIdx.x & 31;
  const int h    = wid % NH;
  const int bs   = wid / NH;
  const int seq  = bs % Sn;
  const int b    = bs / Sn;

  float4 x = *(const float4*)(src + (size_t)wid * HDn + lane * 4);
  float v0 = x.x, v1 = x.y, v2 = x.z, v3 = x.w;

  if (RMS) {
    float ss = v0 * v0 + v1 * v1 + v2 * v2 + v3 * v3;
    #pragma unroll
    for (int m = 1; m < 32; m <<= 1) ss += __shfl_xor(ss, m, 32);
    const float inv = rsqrtf(ss * (1.0f / HDn) + 1e-6f);
    v0 *= inv * scale[lane * 4 + 0];
    v1 *= inv * scale[lane * 4 + 1];
    v2 *= inv * scale[lane * 4 + 2];
    v3 *= inv * scale[lane * 4 + 3];
  }
  if (ROPE) {
    const float fs = (float)seq;
    float ang0 = fs * __powf(1000.0f, -((float)(lane * 4 + 0)) / (float)HDn);
    float ang1 = fs * __powf(1000.0f, -((float)(lane * 4 + 2)) / (float)HDn);
    float c0, s0, c1, s1;
    __sincosf(ang0, &s0, &c0);
    __sincosf(ang1, &s1, &c1);
    float n0 = v0 * c0 - v1 * s0, n1 = v1 * c0 + v0 * s0;
    float n2 = v2 * c1 - v3 * s1, n3 = v3 * c1 + v2 * s1;
    v0 = n0; v1 = n1; v2 = n2; v3 = n3;
  }

  if (!TRANS) {
    const size_t o = ((size_t)(b * NH + h) * Sn + seq) * HDn + lane * 4;
    v4bf out; out[0] = (__bf16)v0; out[1] = (__bf16)v1; out[2] = (__bf16)v2; out[3] = (__bf16)v3;
    *(v4bf*)(dst + o) = out;
  } else {
    const size_t base = (size_t)(b * NH + h) * HDn;
    dst[(base + lane * 4 + 0) * Sn + seq] = (__bf16)v0;
    dst[(base + lane * 4 + 1) * Sn + seq] = (__bf16)v1;
    dst[(base + lane * 4 + 2) * Sn + seq] = (__bf16)v2;
    dst[(base + lane * 4 + 3) * Sn + seq] = (__bf16)v3;
  }
}

// ---------------------------------------------------------------------------
// Flash attention: one wave per 16-query tile; 32 keys/iter; causal mask;
// online softmax; P staged through per-wave LDS in A-fragment layout.
// grid = (S/64, H, B), block = 128 (4 waves). Output bf16 [B,S,H*HD].
// ---------------------------------------------------------------------------
__global__ __launch_bounds__(128) void attn_kernel(
    const __bf16* __restrict__ qb,   // [B,H,S,HD]
    const __bf16* __restrict__ kb,   // [B,G,S,HD]
    const __bf16* __restrict__ vbT,  // [B,G,HD,S]
    __bf16* __restrict__ zbuf)       // [B,S,H*HD]
{
  __shared__ __align__(16) __bf16 Plds[4][16 * 32];

  const int lane = threadIdx.x & 31;
  const int w    = threadIdx.x >> 5;
  const int halo = lane >> 4;
  const int lm   = lane & 15;
  const int qtile = blockIdx.x * 4 + w;
  const int qbase = qtile * 16;
  const int h = blockIdx.y, b = blockIdx.z;
  const int g = h >> 2;  // GS = H/G = 4

  const __bf16* Q  = qb  + (size_t)(b * Hn + h) * Sn * HDn;
  const __bf16* Kp = kb  + (size_t)(b * Gn + g) * Sn * HDn;
  const __bf16* Vt = vbT + (size_t)(b * Gn + g) * HDn * Sn;

  v16bf aq[4];
  #pragma unroll
  for (int i = 0; i < 4; ++i) {
    const size_t base = (size_t)(qbase + lm) * HDn + i * 32 + halo * 8;
    v8bf lo = *(const v8bf*)(Q + base);
    v8bf hi = *(const v8bf*)(Q + base + 16);
    aq[i] = cat16(lo, hi);
  }

  float mrow[8], lrow[8];
  v8f acc[8];
  #pragma unroll
  for (int r = 0; r < 8; ++r) { mrow[r] = -3.0e38f; lrow[r] = 0.0f; }
  #pragma unroll
  for (int j = 0; j < 8; ++j) acc[j] = (v8f){};

  const float scl = 0.08838834764831845f;  // 1/sqrt(128)
  const int kend = qbase + 16;

  for (int kb0 = 0; kb0 < kend; kb0 += 32) {
    v8f sfr[2];
    #pragma unroll
    for (int kt = 0; kt < 2; ++kt) {
      v8f sa = (v8f){};
      const int key = kb0 + kt * 16 + lm;
      #pragma unroll
      for (int i = 0; i < 4; ++i) {
        const size_t base = (size_t)key * HDn + i * 32 + halo * 8;
        v8bf lo = *(const v8bf*)(Kp + base);
        v8bf hi = *(const v8bf*)(Kp + base + 16);
        sa = __builtin_amdgcn_wmma_f32_16x16x32_bf16(
            false, aq[i], false, cat16(lo, hi), (short)0, sa, false, false);
      }
      #pragma unroll
      for (int r = 0; r < 8; ++r) {
        const int rowq = qbase + r + halo * 8;
        sfr[kt][r] = (key > rowq) ? -1.0e30f : sa[r] * scl;
      }
    }
    float corr[8];
    #pragma unroll
    for (int r = 0; r < 8; ++r) {
      float mx = fmaxf(sfr[0][r], sfr[1][r]);
      #pragma unroll
      for (int m = 1; m < 16; m <<= 1) mx = fmaxf(mx, __shfl_xor(mx, m, 32));
      const float nm = fmaxf(mrow[r], mx);
      corr[r] = __expf(mrow[r] - nm);
      mrow[r] = nm;
      const float p0 = __expf(sfr[0][r] - nm);
      const float p1 = __expf(sfr[1][r] - nm);
      sfr[0][r] = p0; sfr[1][r] = p1;
      float sum = p0 + p1;
      #pragma unroll
      for (int m = 1; m < 16; m <<= 1) sum += __shfl_xor(sum, m, 32);
      lrow[r] = lrow[r] * corr[r] + sum;
    }
    #pragma unroll
    for (int j = 0; j < 8; ++j)
      #pragma unroll
      for (int r = 0; r < 8; ++r) acc[j][r] *= corr[r];

    #pragma unroll
    for (int r = 0; r < 8; ++r) {
      const int m = r + halo * 8;
      Plds[w][m * 32 + lm]      = (__bf16)sfr[0][r];
      Plds[w][m * 32 + 16 + lm] = (__bf16)sfr[1][r];
    }
    v8bf plo = *(const v8bf*)&Plds[w][lm * 32 + halo * 8];
    v8bf phi = *(const v8bf*)&Plds[w][lm * 32 + halo * 8 + 16];
    v16bf pf = cat16(plo, phi);

    #pragma unroll
    for (int j = 0; j < 8; ++j) {
      const size_t base = (size_t)(j * 16 + lm) * Sn + kb0 + halo * 8;
      v8bf vlo = *(const v8bf*)(Vt + base);
      v8bf vhi = *(const v8bf*)(Vt + base + 16);
      acc[j] = __builtin_amdgcn_wmma_f32_16x16x32_bf16(
          false, pf, false, cat16(vlo, vhi), (short)0, acc[j], false, false);
    }
  }

  float rinv[8];
  #pragma unroll
  for (int r = 0; r < 8; ++r) rinv[r] = 1.0f / lrow[r];
  #pragma unroll
  for (int j = 0; j < 8; ++j)
    #pragma unroll
    for (int r = 0; r < 8; ++r) {
      const int row = qbase + r + halo * 8;
      const int col = h * HDn + j * 16 + lm;
      zbuf[((size_t)(b * Sn + row)) * (Hn * HDn) + col] = (__bf16)(acc[j][r] * rinv[r]);
    }
}

// ---------------------------------------------------------------------------
// Host side
// ---------------------------------------------------------------------------
extern "C" void kernel_launch(void* const* d_in, const int* in_sizes, int n_in,
                              void* d_out, int out_size, void* d_ws, size_t ws_size,
                              hipStream_t stream) {
  (void)in_sizes; (void)n_in; (void)out_size; (void)ws_size;
  const float* query   = (const float*)d_in[0];
  const float* key     = (const float*)d_in[1];
  const float* value   = (const float*)d_in[2];
  const float* Wq      = (const float*)d_in[3];
  const float* bq      = (const float*)d_in[4];
  const float* Wk      = (const float*)d_in[5];
  const float* bk      = (const float*)d_in[6];
  const float* Wv      = (const float*)d_in[7];
  const float* bv      = (const float*)d_in[8];
  const float* Wo      = (const float*)d_in[9];
  const float* bo      = (const float*)d_in[10];
  const float* q_scale = (const float*)d_in[11];
  // d_in[12] = causal tril mask: applied analytically in attn_kernel.

  size_t off = 0;
  auto take = [&](size_t bytes) -> char* {
    char* p = (char*)d_ws + off;
    off += (bytes + 255) & ~(size_t)255;
    return p;
  };
  const size_t MR = (size_t)Bn * Sn;  // 4096
  // bf16 activations & transposed weights for pure-bf16 GEMMs
  __bf16* qact = (__bf16*)take(MR * Dn * sizeof(__bf16));
  __bf16* kact = (__bf16*)take(MR * Dn * sizeof(__bf16));
  __bf16* vact = (__bf16*)take(MR * Dn * sizeof(__bf16));
  __bf16* WqT  = (__bf16*)take((size_t)Dn * (Hn * HDn) * sizeof(__bf16));
  __bf16* WkT  = (__bf16*)take((size_t)Dn * (Gn * HDn) * sizeof(__bf16));
  __bf16* WvT  = (__bf16*)take((size_t)Dn * (Gn * HDn) * sizeof(__bf16));
  __bf16* WoT  = (__bf16*)take((size_t)(Hn * HDn) * Dn * sizeof(__bf16));
  // f32 projection outputs, bf16 attention operands, bf16 attention output
  float*  qproj = (float*)take(MR * (Hn * HDn) * sizeof(float));
  float*  kproj = (float*)take(MR * (Gn * HDn) * sizeof(float));
  float*  vproj = (float*)take(MR * (Gn * HDn) * sizeof(float));
  __bf16* qbm   = (__bf16*)take(MR * (Hn * HDn) * sizeof(__bf16));
  __bf16* kbm   = (__bf16*)take(MR * (Gn * HDn) * sizeof(__bf16));
  __bf16* vbT   = (__bf16*)take(MR * (Gn * HDn) * sizeof(__bf16));
  __bf16* zbuf  = (__bf16*)take(MR * (Hn * HDn) * sizeof(__bf16));

  // 0) precision prepasses (one-time conversions; ~4us of HBM traffic)
  cvt_bf16_kernel<<<(MR * Dn) / 2048, 256, 0, stream>>>(query, qact);
  cvt_bf16_kernel<<<(MR * Dn) / 2048, 256, 0, stream>>>(key,   kact);
  cvt_bf16_kernel<<<(MR * Dn) / 2048, 256, 0, stream>>>(value, vact);
  wtrans_bf16_kernel<<<((size_t)Dn * (Hn * HDn)) / 1024, 256, 0, stream>>>(Wq, WqT, Hn * HDn, Dn);
  wtrans_bf16_kernel<<<((size_t)Dn * (Gn * HDn)) / 1024, 256, 0, stream>>>(Wk, WkT, Gn * HDn, Dn);
  wtrans_bf16_kernel<<<((size_t)Dn * (Gn * HDn)) / 1024, 256, 0, stream>>>(Wv, WvT, Gn * HDn, Dn);
  wtrans_bf16_kernel<<<((size_t)(Hn * HDn) * Dn) / 1024, 256, 0, stream>>>(Wo, WoT, Dn, Hn * HDn);

  // 1) projections (async-DMA double-buffered bf16 WMMA GEMMs)
  gemm_bf16_kernel<<<dim3((Hn * HDn) / 128, MR / 64), 256, 0, stream>>>(
      qact, WqT, bq, qproj, (int)MR, Hn * HDn, Dn);
  gemm_bf16_kernel<<<dim3((Gn * HDn) / 128, MR / 64), 256, 0, stream>>>(
      kact, WkT, bk, kproj, (int)MR, Gn * HDn, Dn);
  gemm_bf16_kernel<<<dim3((Gn * HDn) / 128, MR / 64), 256, 0, stream>>>(
      vact, WvT, bv, vproj, (int)MR, Gn * HDn, Dn);

  // 2) norm / rope / layout conversion to bf16
  prep_kernel<true,  true,  false><<<(Bn * Sn * Hn) / 8, 256, 0, stream>>>(qproj, qbm, q_scale, Hn);
  prep_kernel<false, true,  false><<<(Bn * Sn * Gn) / 8, 256, 0, stream>>>(kproj, kbm, nullptr, Gn);
  prep_kernel<true,  false, true ><<<(Bn * Sn * Gn) / 8, 256, 0, stream>>>(vproj, vbT, q_scale, Gn);

  // 3) causal flash attention
  attn_kernel<<<dim3(Sn / 64, Hn, Bn), 128, 0, stream>>>(qbm, kbm, vbT, zbuf);

  // 4) output projection
  gemm_bf16_kernel<<<dim3(Dn / 128, MR / 64), 256, 0, stream>>>(
      zbuf, WoT, bo, (float*)d_out, (int)MR, Dn, Hn * HDn);
}
